// XFedFormer_8005819040114
// MI455X (gfx1250) — matrix operand order repaired and checked by reading, not compiled, not run on hardware
//
#include <hip/hip_runtime.h>
#include <hip/hip_bf16.h>
#include <stdint.h>

// ---------------------------------------------------------------------------
// CDNA5 (gfx1250) wave32 WMMA types
// ---------------------------------------------------------------------------
typedef __attribute__((ext_vector_type(16))) __bf16 v16bf;
typedef __attribute__((ext_vector_type(8)))  float  v8f;

union Frag16 { v16bf v; uint32_t u[8]; };

__device__ __forceinline__ uint16_t f2bf(float f) {
  uint32_t u = __float_as_uint(f);
  u += 0x7FFFu + ((u >> 16) & 1u);   // round-to-nearest-even
  return (uint16_t)(u >> 16);
}

// Async global->LDS 16-byte copy (CDNA5 GLOBAL_LOAD_ASYNC_TO_LDS_B128,
// tracked by ASYNCcnt).  LDS address = low 32 bits of the generic pointer.
__device__ __forceinline__ void async_b128(void* lds, const void* g) {
  uint32_t l = (uint32_t)(uintptr_t)lds;
  asm volatile("global_load_async_to_lds_b128 %0, %1, off"
               :: "v"(l), "v"(g) : "memory");
}
__device__ __forceinline__ void wait_async0() {
  asm volatile("s_wait_asynccnt 0x0" ::: "memory");
}

// ---------------------------------------------------------------------------
// Trend decomposition: AvgPool1d(k=7, pad=3, count_include_pad), resid -> bf16
// ---------------------------------------------------------------------------
__launch_bounds__(256)
__global__ void k_decomp(const float* __restrict__ x, uint16_t* __restrict__ residb,
                         float* __restrict__ trend_last, int T, int DIN, int NR) {
  int bt = blockIdx.x;
  int b = bt / T, t = bt % T;
  int d = threadIdx.x;                 // DIN == 256 == blockDim.x
  float s = 0.f;
#pragma unroll
  for (int j = -3; j <= 3; ++j) {
    int tt = t + j;
    if (tt >= 0 && tt < T) s += x[((size_t)b * T + tt) * DIN + d];
  }
  float trend = s * (1.f / 7.f);
  float r = x[((size_t)b * T + t) * DIN + d] - trend;
  residb[((size_t)b * T + t) * DIN + d] = f2bf(r);
  if (t == T - 1 && d < NR) trend_last[(size_t)b * NR + d] = trend;
}

// ---------------------------------------------------------------------------
// fp32 [K][N] -> bf16 [N][K] transposed weight conversion
//   grid = (K/256, N)
// ---------------------------------------------------------------------------
__launch_bounds__(256)
__global__ void k_cvt_t(const float* __restrict__ src, uint16_t* __restrict__ dst,
                        int K, int N) {
  int n = blockIdx.y;
  int k = blockIdx.x * 256 + threadIdx.x;
  dst[(size_t)n * K + k] = f2bf(src[(size_t)k * N + n]);
}

// ---------------------------------------------------------------------------
// Generic bf16 WMMA GEMM: C = epilogue(A[MxK] * B[KxN]), B stored [N][K].
//   block tile 128x128, K-step 32, 8 waves (2M x 4N), wave tile 64x32
//   double-buffered LDS, tiles staged with async global->LDS B128 copies
//   epilogue: +bias, +pos_enc[(row%T)*N+col], act(0=none,1=gelu,2=relu),
//             +resid, * scale[row*sstride] (+= outF if sacc), -> outF/outB
// ---------------------------------------------------------------------------
#define GBM 128
#define GBN 128
#define GBK 32

__device__ __forceinline__ void stage_gemm_tile(
    uint16_t (&dA)[GBM][GBK], uint16_t (&dB)[GBN][GBK],
    const uint16_t* __restrict__ A, const uint16_t* __restrict__ Bw,
    int bm0, int bn0, int Kd, int kk, int tid) {
  // A: 128 rows x 64B -> 512 x B128 chunks; B (already [N][K]): same shape
  for (int c = tid; c < GBM * 4; c += 256) {
    int r = c >> 2, seg = (c & 3) * 8;
    async_b128(&dA[r][seg], A + (size_t)(bm0 + r) * Kd + kk + seg);
  }
  for (int c = tid; c < GBN * 4; c += 256) {
    int r = c >> 2, seg = (c & 3) * 8;
    async_b128(&dB[r][seg], Bw + (size_t)(bn0 + r) * Kd + kk + seg);
  }
}

__launch_bounds__(256)
__global__ void k_gemm(const uint16_t* __restrict__ A, const uint16_t* __restrict__ Bw,
                       const float* __restrict__ bias, const float* __restrict__ resid,
                       const float* __restrict__ pos, int posT,
                       const float* __restrict__ scale, int sstride, int sacc,
                       float* __restrict__ outF, uint16_t* __restrict__ outB,
                       int M, int N, int Kd, int act) {
  __shared__ __align__(16) uint16_t lA[2][GBM][GBK];   // [m][k]
  __shared__ __align__(16) uint16_t lB[2][GBN][GBK];   // [n][k]

  int tid  = threadIdx.x;
  int lane = tid & 31;
  int wave = tid >> 5;
  int half = lane >> 4;
  int l15  = lane & 15;
  int wm0  = (wave & 1) * 64;         // wave M offset inside block tile
  int wn0  = (wave >> 1) * 32;        // wave N offset inside block tile
  int bm0  = blockIdx.y * GBM;
  int bn0  = blockIdx.x * GBN;

  v8f acc[4][2];
#pragma unroll
  for (int i = 0; i < 4; ++i)
#pragma unroll
    for (int j = 0; j < 2; ++j)
#pragma unroll
      for (int r = 0; r < 8; ++r) acc[i][j][r] = 0.f;

  stage_gemm_tile(lA[0], lB[0], A, Bw, bm0, bn0, Kd, 0, tid);

  int nsteps = Kd / GBK;
  for (int s = 0; s < nsteps; ++s) {
    int cur = s & 1;
    wait_async0();          // our async copies for buffer `cur` done
    __syncthreads();        // everyone's done (and prior reads drained)
    if (s + 1 < nsteps)
      stage_gemm_tile(lA[cur ^ 1], lB[cur ^ 1], A, Bw, bm0, bn0, Kd,
                      (s + 1) * GBK, tid);

    Frag16 bfr[2];
#pragma unroll
    for (int nf = 0; nf < 2; ++nf) {
      int col = wn0 + nf * 16 + l15;
      int kb  = half * 16;             // contiguous k order (as A): lanes0-15 K0..15, lanes16-31 K16..31
#pragma unroll
      for (int j = 0; j < 8; ++j)
        bfr[nf].u[j] = *(const uint32_t*)(&lB[cur][col][kb + 2 * j]);
    }
#pragma unroll
    for (int mf = 0; mf < 4; ++mf) {
      Frag16 af;
      int row = wm0 + mf * 16 + l15;
      int kb  = half * 16;             // contiguous k order (as B): lanes0-15 K0..15, lanes16-31 K16..31
#pragma unroll
      for (int j = 0; j < 8; ++j) {
        int k = kb + 2 * j;
        af.u[j] = *(const uint32_t*)(&lA[cur][row][k]);
      }
#pragma unroll
      for (int nf = 0; nf < 2; ++nf)
        acc[mf][nf] = __builtin_amdgcn_wmma_f32_16x16x32_bf16(
            false, af.v, false, bfr[nf].v, (short)0, acc[mf][nf], false, false);
    }
  }

  // epilogue
#pragma unroll
  for (int mf = 0; mf < 4; ++mf)
#pragma unroll
    for (int nf = 0; nf < 2; ++nf) {
      int col = bn0 + wn0 + nf * 16 + l15;
      float bv = bias ? bias[col] : 0.f;
#pragma unroll
      for (int r = 0; r < 8; ++r) {
        int row = bm0 + wm0 + mf * 16 + r + 8 * half;
        float xv = acc[mf][nf][r] + bv;
        if (pos) xv += pos[(size_t)(row % posT) * N + col];
        if (act == 1) xv = 0.5f * xv * (1.f + erff(xv * 0.70710678118f));
        else if (act == 2) xv = fmaxf(xv, 0.f);
        if (resid) xv += resid[(size_t)row * N + col];
        if (scale) {
          xv *= scale[(size_t)row * sstride];
          if (sacc) xv += outF[(size_t)row * N + col];
        }
        if (outF) outF[(size_t)row * N + col] = xv;
        if (outB) outB[(size_t)row * N + col] = f2bf(xv);
      }
    }
}

// ---------------------------------------------------------------------------
// Flash attention (bf16 WMMA, fp32 online softmax).
//   grid = (T/64, H, B), block = 128 (4 waves, 16 queries each)
//   qkv: bf16 [B*T, 3*H*DH] (q | k | v); out: bf16 [B*T, H*DH]
// ---------------------------------------------------------------------------
__launch_bounds__(128)
__global__ void k_attn(const uint16_t* __restrict__ qkv, uint16_t* __restrict__ outB,
                       int T, int Hn, int DHn) {
  __shared__ __align__(16) uint16_t lK[32][64];   // [key][d]   (async-staged)
  __shared__ uint16_t lVt[64][32];                // [d][key]   (transposed at fill)
  __shared__ uint16_t lP[4][16][32];              // per-wave P staging

  int tid = threadIdx.x, lane = tid & 31, wave = tid >> 5;
  int half = lane >> 4, l15 = lane & 15;
  int b = blockIdx.z, h = blockIdx.y;
  int q0 = blockIdx.x * 64 + wave * 16;
  int strideQ = 3 * Hn * DHn;
  size_t baseQ = (size_t)(b * T + q0) * strideQ + h * DHn;
  size_t baseK = (size_t)(b * T) * strideQ + Hn * DHn + h * DHn;
  size_t baseV = (size_t)(b * T) * strideQ + 2 * Hn * DHn + h * DHn;
  float sc = 1.f / sqrtf((float)DHn);

  // Q fragments: A-layout, two K=32 chunks of DH=64
  Frag16 qf[2];
  {
    int kb = half * 16;
#pragma unroll
    for (int f = 0; f < 2; ++f)
#pragma unroll
      for (int j = 0; j < 8; ++j) {
        int d = f * 32 + kb + 2 * j;
        qf[f].u[j] = *(const uint32_t*)(qkv + baseQ + (size_t)l15 * strideQ + d);
      }
  }

  float mrow[8], lrow[8];
  v8f o[4];
#pragma unroll
  for (int r = 0; r < 8; ++r) { mrow[r] = -1e30f; lrow[r] = 0.f; }
#pragma unroll
  for (int nf = 0; nf < 4; ++nf)
#pragma unroll
    for (int r = 0; r < 8; ++r) o[nf][r] = 0.f;

  for (int kb0 = 0; kb0 < T; kb0 += 32) {
    __syncthreads();
    // K tile: async B128 copies (32 keys x 128B = 256 chunks, 2 per thread)
    for (int c = tid; c < 256; c += 128) {
      int key = c >> 3, seg = (c & 7) * 8;
      async_b128(&lK[key][seg], qkv + baseK + (size_t)(kb0 + key) * strideQ + seg);
    }
    // V tile: transposed 16-bit scatter (sync)
    for (int i = tid; i < (32 * 64) / 2; i += 128) {
      int key = i >> 5;
      int dp  = (i & 31) * 2;
      uint32_t vv = *(const uint32_t*)(qkv + baseV + (size_t)(kb0 + key) * strideQ + dp);
      lVt[dp][key]     = (uint16_t)(vv & 0xffffu);
      lVt[dp + 1][key] = (uint16_t)(vv >> 16);
    }
    wait_async0();
    __syncthreads();

    // S = Q K^T : two 16-key column tiles, reduce over DH in 2 WMMAs each
    v8f s[2];
#pragma unroll
    for (int nk = 0; nk < 2; ++nk) {
#pragma unroll
      for (int r = 0; r < 8; ++r) s[nk][r] = 0.f;
      int key = nk * 16 + l15;
#pragma unroll
      for (int ds = 0; ds < 2; ++ds) {
        Frag16 bf;
        int db = ds * 32 + half * 16;
#pragma unroll
        for (int j = 0; j < 8; ++j)
          bf.u[j] = *(const uint32_t*)(&lK[key][db + 2 * j]);
        s[nk] = __builtin_amdgcn_wmma_f32_16x16x32_bf16(
            false, qf[ds].v, false, bf.v, (short)0, s[nk], false, false);
      }
#pragma unroll
      for (int r = 0; r < 8; ++r) s[nk][r] *= sc;
    }

    // online softmax over the 32 new keys (row = r + 8*half)
    float corr[8];
#pragma unroll
    for (int r = 0; r < 8; ++r) {
      float v = fmaxf(s[0][r], s[1][r]);
#pragma unroll
      for (int off = 1; off < 16; off <<= 1) v = fmaxf(v, __shfl_xor(v, off, 32));
      float mn = fmaxf(mrow[r], v);
      corr[r] = __expf(mrow[r] - mn);
      mrow[r] = mn;
      float p0 = __expf(s[0][r] - mn);
      float p1 = __expf(s[1][r] - mn);
      s[0][r] = p0; s[1][r] = p1;
      float t = p0 + p1;
#pragma unroll
      for (int off = 1; off < 16; off <<= 1) t += __shfl_xor(t, off, 32);
      lrow[r] = lrow[r] * corr[r] + t;
    }
#pragma unroll
    for (int nf = 0; nf < 4; ++nf)
#pragma unroll
      for (int r = 0; r < 8; ++r) o[nf][r] *= corr[r];

    // P: C-layout -> LDS -> A-layout for the AV WMMA
#pragma unroll
    for (int nk = 0; nk < 2; ++nk)
#pragma unroll
      for (int r = 0; r < 8; ++r)
        lP[wave][r + 8 * half][nk * 16 + l15] = f2bf(s[nk][r]);
    __syncthreads();

    Frag16 pa;
    {
      int kb = half * 16;
#pragma unroll
      for (int j = 0; j < 8; ++j) {
        int k = kb + 2 * j;
        pa.u[j] = *(const uint32_t*)(&lP[wave][l15][k]);
      }
    }
#pragma unroll
    for (int nf = 0; nf < 4; ++nf) {
      Frag16 vb;
      int d = nf * 16 + l15;
      int keyb = half * 16;
#pragma unroll
      for (int j = 0; j < 8; ++j)
        vb.u[j] = *(const uint32_t*)(&lVt[d][keyb + 2 * j]);
      o[nf] = __builtin_amdgcn_wmma_f32_16x16x32_bf16(
          false, pa.v, false, vb.v, (short)0, o[nf], false, false);
    }
  }

  // finalize + store bf16
  int Dm = Hn * DHn;
#pragma unroll
  for (int nf = 0; nf < 4; ++nf)
#pragma unroll
    for (int r = 0; r < 8; ++r) {
      size_t row = (size_t)(b * T + q0 + r + 8 * half);
      outB[row * Dm + h * DHn + nf * 16 + l15] = f2bf(o[nf][r] / lrow[r]);
    }
}

// ---------------------------------------------------------------------------
// LayerNorm: out = (inp[+resid] - mean)/sqrt(var+eps)*g + b  (D == 512)
// ---------------------------------------------------------------------------
__launch_bounds__(256)
__global__ void k_ln(const float* __restrict__ inp, const float* __restrict__ resid,
                     const float* __restrict__ g, const float* __restrict__ be,
                     float* __restrict__ outF, uint16_t* __restrict__ outB, int D) {
  int row = blockIdx.x;
  int tid = threadIdx.x;
  size_t base = (size_t)row * D;
  float v0 = inp[base + tid];
  float v1 = inp[base + tid + 256];
  if (resid) { v0 += resid[base + tid]; v1 += resid[base + tid + 256]; }
  float s = v0 + v1, ss = v0 * v0 + v1 * v1;
#pragma unroll
  for (int off = 16; off; off >>= 1) {
    s  += __shfl_xor(s, off, 32);
    ss += __shfl_xor(ss, off, 32);
  }
  __shared__ float red[2][8];
  if ((tid & 31) == 0) { red[0][tid >> 5] = s; red[1][tid >> 5] = ss; }
  __syncthreads();
  float S = 0.f, SS = 0.f;
#pragma unroll
  for (int i = 0; i < 8; ++i) { S += red[0][i]; SS += red[1][i]; }
  float mean = S / (float)D;
  float var  = SS / (float)D - mean * mean;
  float inv  = rsqrtf(var + 1e-5f);
  float o0 = (v0 - mean) * inv * g[tid]       + be[tid];
  float o1 = (v1 - mean) * inv * g[tid + 256] + be[tid + 256];
  if (outF) { outF[base + tid] = o0; outF[base + tid + 256] = o1; }
  if (outB) { outB[base + tid] = f2bf(o0); outB[base + tid + 256] = f2bf(o1); }
}

// ---------------------------------------------------------------------------
// MoE gating: logits = z@Wg + bg; top-2 of 4; softmax over top-2 -> comb[M,4]
// ---------------------------------------------------------------------------
__launch_bounds__(256)
__global__ void k_gate(const float* __restrict__ z, const float* __restrict__ Wg,
                       const float* __restrict__ bg, float* __restrict__ comb, int D) {
  int wave = threadIdx.x >> 5, lane = threadIdx.x & 31;
  int row = blockIdx.x * 8 + wave;
  float a[4] = {0.f, 0.f, 0.f, 0.f};
  for (int d = lane; d < D; d += 32) {
    float zv = z[(size_t)row * D + d];
#pragma unroll
    for (int e = 0; e < 4; ++e) a[e] += zv * Wg[d * 4 + e];
  }
#pragma unroll
  for (int off = 16; off; off >>= 1)
#pragma unroll
    for (int e = 0; e < 4; ++e) a[e] += __shfl_xor(a[e], off, 32);
  if (lane == 0) {
#pragma unroll
    for (int e = 0; e < 4; ++e) a[e] += bg[e];
    int i1 = 0;
#pragma unroll
    for (int e = 1; e < 4; ++e) if (a[e] > a[i1]) i1 = e;   // first index wins ties
    int i2 = -1;
#pragma unroll
    for (int e = 0; e < 4; ++e)
      if (e != i1 && (i2 < 0 || a[e] > a[i2])) i2 = e;
    float ex = __expf(a[i2] - a[i1]);
    float w1 = 1.f / (1.f + ex);
    float w2 = ex * w1;
#pragma unroll
    for (int e = 0; e < 4; ++e) comb[(size_t)row * 4 + e] = 0.f;
    comb[(size_t)row * 4 + i1] = w1;
    comb[(size_t)row * 4 + i2] = w2;
  }
}

// ---------------------------------------------------------------------------
// Decoder: out[b,n] = z[b, T-1, :]@Wd[:,n] + bd[n] + trend_last[b,n]
// ---------------------------------------------------------------------------
__launch_bounds__(256)
__global__ void k_dec(const float* __restrict__ z, const float* __restrict__ Wd,
                      const float* __restrict__ bd, const float* __restrict__ tl,
                      float* __restrict__ out, int T, int D, int NR) {
  int b = blockIdx.x, n = threadIdx.x;
  const float* zr = z + ((size_t)b * T + T - 1) * D;
  float s = 0.f;
  for (int d = 0; d < D; ++d) s = fmaf(zr[d], Wd[(size_t)d * NR + n], s);
  out[(size_t)b * NR + n] = s + bd[n] + tl[(size_t)b * NR + n];
}

// ---------------------------------------------------------------------------
// host: full pipeline
// ---------------------------------------------------------------------------
extern "C" void kernel_launch(void* const* d_in, const int* in_sizes, int n_in,
                              void* d_out, int out_size, void* d_ws, size_t ws_size,
                              hipStream_t stream) {
  (void)in_sizes; (void)n_in; (void)out_size; (void)ws_size;
  constexpr int B = 32, T = 512, DIN = 256, NR = 256;
  constexpr int D = 512, H = 8, L = 3, DH = 64, FF = 2048, EH = 1024;
  constexpr int M = B * T;

  const float* x_series = (const float*)d_in[0];
  const float* pos_enc  = (const float*)d_in[1];
  const float* W_in  = (const float*)d_in[2];
  const float* b_in  = (const float*)d_in[3];
  const float* Wqkv  = (const float*)d_in[4];
  const float* bqkv  = (const float*)d_in[5];
  const float* Wo    = (const float*)d_in[6];
  const float* bo    = (const float*)d_in[7];
  const float* ln1_g = (const float*)d_in[8];
  const float* ln1_b = (const float*)d_in[9];
  const float* W1    = (const float*)d_in[10];
  const float* b1    = (const float*)d_in[11];
  const float* W2    = (const float*)d_in[12];
  const float* b2    = (const float*)d_in[13];
  const float* ln2_g = (const float*)d_in[14];
  const float* ln2_b = (const float*)d_in[15];
  const float* lnf_g = (const float*)d_in[16];
  const float* lnf_b = (const float*)d_in[17];
  const float* Wg    = (const float*)d_in[18];
  const float* bg    = (const float*)d_in[19];
  const float* We1   = (const float*)d_in[20];
  const float* be1   = (const float*)d_in[21];
  const float* We2   = (const float*)d_in[22];
  const float* be2   = (const float*)d_in[23];
  const float* lnm_g = (const float*)d_in[24];
  const float* lnm_b = (const float*)d_in[25];
  const float* Wd    = (const float*)d_in[26];
  const float* bd    = (const float*)d_in[27];

  uint8_t* wsp = (uint8_t*)d_ws;
  size_t off = 0;
  auto alloc = [&](size_t bytes) -> void* {
    void* p = wsp + off;
    off += (bytes + 255) & ~(size_t)255;
    return p;
  };
  // bf16 weights (all stored transposed: [N][K])
  uint16_t* wWin  = (uint16_t*)alloc((size_t)DIN * D * 2);
  uint16_t* wWqkv = (uint16_t*)alloc((size_t)L * D * 3 * D * 2);
  uint16_t* wWo   = (uint16_t*)alloc((size_t)L * D * D * 2);
  uint16_t* wW1   = (uint16_t*)alloc((size_t)L * D * FF * 2);
  uint16_t* wW2   = (uint16_t*)alloc((size_t)L * FF * D * 2);
  uint16_t* wWe1  = (uint16_t*)alloc((size_t)4 * D * EH * 2);
  uint16_t* wWe2  = (uint16_t*)alloc((size_t)4 * EH * D * 2);
  // activations
  uint16_t* residb = (uint16_t*)alloc((size_t)M * DIN * 2);
  float*    trendl = (float*)   alloc((size_t)B * NR * 4);
  float*    comb   = (float*)   alloc((size_t)M * 4 * 4);
  float*    zf     = (float*)   alloc((size_t)M * D * 4);
  uint16_t* zb     = (uint16_t*)alloc((size_t)M * D * 2);
  uint16_t* bigb   = (uint16_t*)alloc((size_t)M * FF * 2);   // qkv / ff1 / expert-h
  uint16_t* attb   = (uint16_t*)alloc((size_t)M * D * 2);
  float*    scrF   = (float*)   alloc((size_t)M * D * 4);

  auto cvtT = [&](const float* s, uint16_t* d, int K, int N) {
    dim3 g(K / 256, N);
    k_cvt_t<<<g, 256, 0, stream>>>(s, d, K, N);
  };
  auto gemm = [&](const uint16_t* A, const uint16_t* Bw, const float* bias,
                  const float* resid, const float* pos,
                  const float* scale, int sstride, int sacc,
                  float* outF, uint16_t* outB, int Mm, int Nn, int Kd, int act) {
    dim3 g(Nn / GBN, Mm / GBM);
    k_gemm<<<g, 256, 0, stream>>>(A, Bw, bias, resid, pos, T, scale, sstride, sacc,
                                  outF, outB, Mm, Nn, Kd, act);
  };

  // 0) decomposition + transposed weight conversion
  k_decomp<<<B * T, 256, 0, stream>>>(x_series, residb, trendl, T, DIN, NR);
  cvtT(W_in, wWin, DIN, D);
  for (int l = 0; l < L; ++l) {
    cvtT(Wqkv + (size_t)l * D * 3 * D, wWqkv + (size_t)l * D * 3 * D, D, 3 * D);
    cvtT(Wo   + (size_t)l * D * D,     wWo   + (size_t)l * D * D,     D, D);
    cvtT(W1   + (size_t)l * D * FF,    wW1   + (size_t)l * D * FF,    D, FF);
    cvtT(W2   + (size_t)l * FF * D,    wW2   + (size_t)l * FF * D,    FF, D);
  }
  for (int e = 0; e < 4; ++e) {
    cvtT(We1 + (size_t)e * D * EH, wWe1 + (size_t)e * D * EH, D, EH);
    cvtT(We2 + (size_t)e * EH * D, wWe2 + (size_t)e * EH * D, EH, D);
  }

  // 1) input projection + positional encoding
  gemm(residb, wWin, b_in, nullptr, pos_enc, nullptr, 0, 0, zf, zb, M, D, DIN, 0);

  // 2) transformer encoder layers
  for (int l = 0; l < L; ++l) {
    gemm(zb, wWqkv + (size_t)l * D * 3 * D, bqkv + (size_t)l * 3 * D,
         nullptr, nullptr, nullptr, 0, 0, nullptr, bigb, M, 3 * D, D, 0);
    k_attn<<<dim3(T / 64, H, B), 128, 0, stream>>>(bigb, attb, T, H, DH);
    gemm(attb, wWo + (size_t)l * D * D, bo + (size_t)l * D,
         zf, nullptr, nullptr, 0, 0, scrF, nullptr, M, D, D, 0);
    k_ln<<<M, 256, 0, stream>>>(scrF, nullptr, ln1_g + (size_t)l * D, ln1_b + (size_t)l * D,
                                zf, zb, D);
    gemm(zb, wW1 + (size_t)l * D * FF, b1 + (size_t)l * FF,
         nullptr, nullptr, nullptr, 0, 0, nullptr, bigb, M, FF, D, 1 /*gelu*/);
    gemm(bigb, wW2 + (size_t)l * FF * D, b2 + (size_t)l * D,
         zf, nullptr, nullptr, 0, 0, scrF, nullptr, M, D, FF, 0);
    k_ln<<<M, 256, 0, stream>>>(scrF, nullptr, ln2_g + (size_t)l * D, ln2_b + (size_t)l * D,
                                zf, zb, D);
  }
  // encoder final norm
  k_ln<<<M, 256, 0, stream>>>(zf, nullptr, lnf_g, lnf_b, zf, zb, D);

  // 3) MoE: gating + dense-equivalent expert mixing
  k_gate<<<M / 8, 256, 0, stream>>>(zf, Wg, bg, comb, D);
  for (int e = 0; e < 4; ++e) {
    gemm(zb, wWe1 + (size_t)e * D * EH, be1 + (size_t)e * EH,
         nullptr, nullptr, nullptr, 0, 0, nullptr, bigb, M, EH, D, 2 /*relu*/);
    gemm(bigb, wWe2 + (size_t)e * EH * D, be2 + (size_t)e * D,
         nullptr, nullptr, comb + e, 4, (e > 0) ? 1 : 0, scrF, nullptr, M, D, EH, 0);
  }
  k_ln<<<M, 256, 0, stream>>>(scrF, zf, lnm_g, lnm_b, zf, nullptr, D);

  // 4) decoder + trend add
  k_dec<<<B, 256, 0, stream>>>(zf, Wd, bd, trendl, (float*)d_out, T, D, NR);
}
